// Model_36223754174773
// MI455X (gfx1250) — matrix operand order repaired and checked by reading, not compiled, not run on hardware
//
#include <hip/hip_runtime.h>
#include <hip/hip_bf16.h>
#include <math.h>

#define NB     50000
#define NEDGE  800000
#define ETOT   (NEDGE + NB)
#define FDIM   128
#define D1     192
#define NHEADS 6
#define C1     32
#define NCLS   10
#define N2PAD  16
#define ENC_NEG_INF 0x007FFFFFu

typedef __attribute__((ext_vector_type(16))) __bf16 v16bf;
typedef __attribute__((ext_vector_type(8)))  __bf16 v8bf;
typedef __attribute__((ext_vector_type(8)))  float  v8f;

// ---------- helpers ----------
__device__ __forceinline__ unsigned f2ord(float f) {
    unsigned u = __float_as_uint(f);
    return (u & 0x80000000u) ? ~u : (u | 0x80000000u);
}
__device__ __forceinline__ float ord2f(unsigned u) {
    unsigned v = (u & 0x80000000u) ? (u & 0x7FFFFFFFu) : ~u;
    return __uint_as_float(v);
}
__device__ __forceinline__ void edge_ends(const long long* __restrict__ ei, int e, int& s, int& d) {
    if (e < NEDGE) { s = (int)ei[e]; d = (int)ei[NEDGE + e]; }
    else           { s = e - NEDGE; d = e - NEDGE; }
}
__device__ __forceinline__ float lrelu(float t) { return (t > 0.0f) ? t : 0.2f * t; }

// ---------- fills / conversions ----------
__global__ void fill_u32(unsigned* p, unsigned v, int n) {
    int i = blockIdx.x * blockDim.x + threadIdx.x;
    if (i < n) p[i] = v;
}
__global__ void fill_f32(float* p, float v, int n) {
    int i = blockIdx.x * blockDim.x + threadIdx.x;
    if (i < n) p[i] = v;
}
__global__ void cvt_bf16(const float* __restrict__ in, __bf16* __restrict__ out, int n) {
    int i = blockIdx.x * blockDim.x + threadIdx.x;
    if (i < n) out[i] = (__bf16)in[i];
}
// pad rows with zeros (layer-2 weights: 10 real rows -> 16)
__global__ void cvt_bf16_pad(const float* __restrict__ in, __bf16* __restrict__ out,
                             int rows, int padRows, int cols) {
    int i = blockIdx.x * blockDim.x + threadIdx.x;
    int tot = padRows * cols;
    if (i >= tot) return;
    int r = i / cols, c = i % cols;
    out[i] = (r < rows) ? (__bf16)in[r * cols + c] : (__bf16)0.0f;
}

// ---------- WMMA bf16 GEMM:  out[M,N] = A[M,K] * B[N,K]^T + bias ----------
__global__ __launch_bounds__(256) void gemm_wmma_bf16(
    const __bf16* __restrict__ A,   // M x K row-major (bf16)
    const __bf16* __restrict__ W,   // N x K row-major (bf16)  (B = W^T)
    const float*  __restrict__ bias, int biasN,
    float* __restrict__ out,        // M x N row-major (f32)
    int M, int K, int N, int ntiles)
{
    int waveId = (blockIdx.x * blockDim.x + threadIdx.x) >> 5;
    int lane   = threadIdx.x & 31;
    int mtile  = waveId / ntiles;
    int ntile  = waveId % ntiles;
    if (mtile * 16 >= M) return;

    int laneM = lane & 15;
    int hi    = (lane >> 4) & 1;

    // A fragment per lane: row = mtile*16+laneM, K = {hi*8+0..7, 16+hi*8+0..7} per step
    const __bf16* aRow = A + (size_t)(mtile * 16 + laneM) * K + hi * 8;
    // B fragment per lane: col = ntile*16+laneM, K = {hi*8+0..7, 16+hi*8+0..7} per step
    const __bf16* bRow = W + (size_t)(ntile * 16 + laneM) * K + hi * 8;

    v8f c = {};
    for (int k = 0; k < K; k += 32) {
        v8bf a0 = *(const v8bf*)(aRow + k);
        v8bf a1 = *(const v8bf*)(aRow + k + 16);
        v8bf b0 = *(const v8bf*)(bRow + k);
        v8bf b1 = *(const v8bf*)(bRow + k + 16);
        v16bf av = __builtin_shufflevector(a0, a1, 0,1,2,3,4,5,6,7,8,9,10,11,12,13,14,15);
        v16bf bv = __builtin_shufflevector(b0, b1, 0,1,2,3,4,5,6,7,8,9,10,11,12,13,14,15);
        c = __builtin_amdgcn_wmma_f32_16x16x32_bf16(
                /*neg_a=*/false, av, /*neg_b=*/false, bv,
                /*c_mod=*/(short)0, c, /*reuse_a=*/false, /*reuse_b=*/false);
    }

    int col = ntile * 16 + laneM;
    float bv = (col < biasN) ? bias[col] : 0.0f;
#pragma unroll
    for (int i = 0; i < 8; ++i) {
        int row = mtile * 16 + hi * 8 + i;
        out[(size_t)row * N + col] = c[i] + bv;
    }
}

// ---------- edge pass A: score + segment-max (float4 vectorized) ----------
template <int HEADS, int C, int LD>
__global__ void edge_score_t(const long long* __restrict__ ei,
                             const float* __restrict__ xl, const float* __restrict__ xr,
                             const float* __restrict__ att,
                             float* __restrict__ score, unsigned* __restrict__ smax)
{
    int idx = blockIdx.x * blockDim.x + threadIdx.x;
    if (idx >= ETOT * HEADS) return;
    int e = idx / HEADS, h = idx % HEADS;
    int s, d; edge_ends(ei, e, s, d);
    const float* pl = xl + (size_t)s * LD + h * C;
    const float* pr = xr + (size_t)d * LD + h * C;
    const float* pa = att + h * C;
    const float4* pl4 = reinterpret_cast<const float4*>(pl);
    const float4* pr4 = reinterpret_cast<const float4*>(pr);
    const float4* pa4 = reinterpret_cast<const float4*>(pa);
    float acc = 0.0f;
    constexpr int C4 = C / 4;
#pragma unroll
    for (int c = 0; c < C4; ++c) {
        float4 a = pl4[c], b = pr4[c], w = pa4[c];
        acc += w.x * lrelu(a.x + b.x);
        acc += w.y * lrelu(a.y + b.y);
        acc += w.z * lrelu(a.z + b.z);
        acc += w.w * lrelu(a.w + b.w);
    }
#pragma unroll
    for (int c = C4 * 4; c < C; ++c)
        acc += pa[c] * lrelu(pl[c] + pr[c]);
    score[idx] = acc;
    atomicMax(smax + (size_t)d * HEADS + h, f2ord(acc));
}

// ---------- edge pass B: exp + segment-sum ----------
template <int HEADS>
__global__ void edge_exp_t(const long long* __restrict__ ei,
                           float* __restrict__ score, const unsigned* __restrict__ smax,
                           float* __restrict__ denom)
{
    int idx = blockIdx.x * blockDim.x + threadIdx.x;
    if (idx >= ETOT * HEADS) return;
    int e = idx / HEADS, h = idx % HEADS;
    int s, d; edge_ends(ei, e, s, d);
    float m  = ord2f(smax[(size_t)d * HEADS + h]);
    float ex = expf(score[idx] - m);
    score[idx] = ex;
    atomicAdd(denom + (size_t)d * HEADS + h, ex);
}

// ---------- edge pass C: alpha-weighted scatter-add (float4 loads) ----------
template <int HEADS, int C, int LD, int LDO>
__global__ void edge_agg_t(const long long* __restrict__ ei,
                           const float* __restrict__ ex, const float* __restrict__ denom,
                           const float* __restrict__ xl, float* __restrict__ agg)
{
    int idx = blockIdx.x * blockDim.x + threadIdx.x;
    if (idx >= ETOT * HEADS) return;
    int e = idx / HEADS, h = idx % HEADS;
    int s, d; edge_ends(ei, e, s, d);
    float alpha = ex[idx] / denom[(size_t)d * HEADS + h];
    const float* pl = xl + (size_t)s * LD + h * C;
    float* po = agg + (size_t)d * LDO + h * C;
    const float4* pl4 = reinterpret_cast<const float4*>(pl);
    constexpr int C4 = C / 4;
#pragma unroll
    for (int c = 0; c < C4; ++c) {
        float4 a = pl4[c];
        atomicAdd(po + 4 * c + 0, a.x * alpha);
        atomicAdd(po + 4 * c + 1, a.y * alpha);
        atomicAdd(po + 4 * c + 2, a.z * alpha);
        atomicAdd(po + 4 * c + 3, a.w * alpha);
    }
#pragma unroll
    for (int c = C4 * 4; c < C; ++c)
        atomicAdd(po + c, pl[c] * alpha);
}

// ---------- node: h = elu(agg + b1), plus bf16 copy for layer-2 GEMM ----------
__global__ void node_elu(float* __restrict__ agg, const float* __restrict__ bias,
                         __bf16* __restrict__ hbf, int n, int ld)
{
    int i = blockIdx.x * blockDim.x + threadIdx.x;
    if (i >= n) return;
    int col = i % ld;
    float v = agg[i] + bias[col];
    v = (v > 0.0f) ? v : (expf(v) - 1.0f);      // elu
    agg[i] = v;
    hbf[i] = (__bf16)v;
}

// ---------- node: v[n] = mean_c (out2[n,c] + b2[c]) ----------
__global__ void node_meanv(const float* __restrict__ out2, const float* __restrict__ b2,
                           float* __restrict__ v, int n)
{
    int i = blockIdx.x * blockDim.x + threadIdx.x;
    if (i >= n) return;
    const float4* r4 = reinterpret_cast<const float4*>(out2 + (size_t)i * N2PAD);
    float4 r0 = r4[0], r1 = r4[1];
    float acc = r0.x + r0.y + r0.z + r0.w + r1.x + r1.y + r1.z + r1.w
              + out2[(size_t)i * N2PAD + 8] + out2[(size_t)i * N2PAD + 9];
#pragma unroll
    for (int c = 0; c < NCLS; ++c) acc += b2[c];
    v[i] = acc * (1.0f / NCLS);
}

// ---------- global log-softmax over all nodes ----------
__global__ void reduce_max(const float* __restrict__ v, unsigned* gmax, int n) {
    int i = blockIdx.x * blockDim.x + threadIdx.x;
    if (i < n) atomicMax(gmax, f2ord(v[i]));
}
__global__ void reduce_sumexp(const float* __restrict__ v, const unsigned* gmax,
                              float* gsum, int n) {
    int i = blockIdx.x * blockDim.x + threadIdx.x;
    if (i < n) atomicAdd(gsum, expf(v[i] - ord2f(*gmax)));
}
__global__ void finalize_lsm(const float* __restrict__ v, const unsigned* gmax,
                             const float* gsum, float* __restrict__ out, int n) {
    int i = blockIdx.x * blockDim.x + threadIdx.x;
    if (i < n) out[i] = v[i] - ord2f(*gmax) - logf(*gsum);
}

// ---------- driver ----------
extern "C" void kernel_launch(void* const* d_in, const int* in_sizes, int n_in,
                              void* d_out, int out_size, void* d_ws, size_t ws_size,
                              hipStream_t stream) {
    (void)in_sizes; (void)n_in; (void)out_size; (void)ws_size;
    const float*     x    = (const float*)d_in[0];
    const long long* ei   = (const long long*)d_in[1];
    const float*     Wl1  = (const float*)d_in[2];
    const float*     bl1  = (const float*)d_in[3];
    const float*     Wr1  = (const float*)d_in[4];
    const float*     br1  = (const float*)d_in[5];
    const float*     att1 = (const float*)d_in[6];
    const float*     b1   = (const float*)d_in[7];
    const float*     Wl2  = (const float*)d_in[8];
    const float*     bl2  = (const float*)d_in[9];
    const float*     Wr2  = (const float*)d_in[10];
    const float*     br2  = (const float*)d_in[11];
    const float*     att2 = (const float*)d_in[12];
    const float*     b2   = (const float*)d_in[13];
    float* out = (float*)d_out;

    // workspace layout (256B aligned)
    char* ws = (char*)d_ws;
    size_t off = 0;
    auto alloc = [&](size_t bytes) -> char* {
        char* p = ws + off;
        off += (bytes + 255) & ~(size_t)255;
        return p;
    };
    __bf16*   xbf  = (__bf16*)  alloc((size_t)NB * FDIM * 2);
    __bf16*   wl1b = (__bf16*)  alloc((size_t)D1 * FDIM * 2);
    __bf16*   wr1b = (__bf16*)  alloc((size_t)D1 * FDIM * 2);
    float*    xl1  = (float*)   alloc((size_t)NB * D1 * 4);
    float*    xr1  = (float*)   alloc((size_t)NB * D1 * 4);
    float*    sc1  = (float*)   alloc((size_t)ETOT * NHEADS * 4);
    unsigned* sm1  = (unsigned*)alloc((size_t)NB * NHEADS * 4);
    float*    dn1  = (float*)   alloc((size_t)NB * NHEADS * 4);
    float*    agg1 = (float*)   alloc((size_t)NB * D1 * 4);
    __bf16*   hbf  = (__bf16*)  alloc((size_t)NB * D1 * 2);
    // layer-2 buffers alias xl1's region (xl1/xr1 dead after edge_agg layer 1)
    char* l2 = (char*)xl1;
    size_t o2 = 0;
    auto alloc2 = [&](size_t bytes) -> char* {
        char* p = l2 + o2;
        o2 += (bytes + 255) & ~(size_t)255;
        return p;
    };
    float*    xl2  = (float*)   alloc2((size_t)NB * N2PAD * 4);
    float*    xr2  = (float*)   alloc2((size_t)NB * N2PAD * 4);
    float*    sc2  = (float*)   alloc2((size_t)ETOT * 4);
    unsigned* sm2  = (unsigned*)alloc2((size_t)NB * 4);
    float*    dn2  = (float*)   alloc2((size_t)NB * 4);
    float*    out2 = (float*)   alloc2((size_t)NB * N2PAD * 4);
    float*    vbuf = (float*)   alloc2((size_t)NB * 4);
    __bf16*   w2lb = (__bf16*)  alloc2((size_t)N2PAD * D1 * 2);
    __bf16*   w2rb = (__bf16*)  alloc2((size_t)N2PAD * D1 * 2);
    unsigned* gmax = (unsigned*)alloc2(256);
    float*    gsum = (float*)   alloc2(256);

    const int B = 256;
    auto nb = [](long long n, int b) { return (int)((n + b - 1) / b); };

    // ---- layer 1 prep ----
    cvt_bf16<<<nb((long long)NB * FDIM, B), B, 0, stream>>>(x,   xbf,  NB * FDIM);
    cvt_bf16<<<nb((long long)D1 * FDIM, B), B, 0, stream>>>(Wl1, wl1b, D1 * FDIM);
    cvt_bf16<<<nb((long long)D1 * FDIM, B), B, 0, stream>>>(Wr1, wr1b, D1 * FDIM);
    fill_u32<<<nb((long long)NB * NHEADS, B), B, 0, stream>>>(sm1, ENC_NEG_INF, NB * NHEADS);
    fill_f32<<<nb((long long)NB * NHEADS, B), B, 0, stream>>>(dn1, 0.0f, NB * NHEADS);
    fill_f32<<<nb((long long)NB * D1, B), B, 0, stream>>>(agg1, 0.0f, NB * D1);

    // ---- layer 1 GEMMs: xl1 = x Wl1^T + bl1 ; xr1 = x Wr1^T + br1 ----
    long long thr1 = (long long)(NB / 16) * (D1 / 16) * 32;
    gemm_wmma_bf16<<<nb(thr1, B), B, 0, stream>>>(xbf, wl1b, bl1, D1, xl1, NB, FDIM, D1, D1 / 16);
    gemm_wmma_bf16<<<nb(thr1, B), B, 0, stream>>>(xbf, wr1b, br1, D1, xr1, NB, FDIM, D1, D1 / 16);

    // ---- layer 1 edge passes ----
    long long ee1 = (long long)ETOT * NHEADS;
    edge_score_t<NHEADS, C1, D1><<<nb(ee1, B), B, 0, stream>>>(ei, xl1, xr1, att1, sc1, sm1);
    edge_exp_t<NHEADS>          <<<nb(ee1, B), B, 0, stream>>>(ei, sc1, sm1, dn1);
    edge_agg_t<NHEADS, C1, D1, D1><<<nb(ee1, B), B, 0, stream>>>(ei, sc1, dn1, xl1, agg1);

    // ---- elu + bf16 copy ----
    node_elu<<<nb((long long)NB * D1, B), B, 0, stream>>>(agg1, b1, hbf, NB * D1, D1);

    // ---- layer 2 prep (xl1 region now reusable) ----
    cvt_bf16_pad<<<nb((long long)N2PAD * D1, B), B, 0, stream>>>(Wl2, w2lb, NCLS, N2PAD, D1);
    cvt_bf16_pad<<<nb((long long)N2PAD * D1, B), B, 0, stream>>>(Wr2, w2rb, NCLS, N2PAD, D1);
    fill_u32<<<nb(NB, B), B, 0, stream>>>(sm2, ENC_NEG_INF, NB);
    fill_f32<<<nb(NB, B), B, 0, stream>>>(dn2, 0.0f, NB);
    fill_f32<<<nb((long long)NB * N2PAD, B), B, 0, stream>>>(out2, 0.0f, NB * N2PAD);
    fill_u32<<<1, 1, 0, stream>>>(gmax, ENC_NEG_INF, 1);
    fill_f32<<<1, 1, 0, stream>>>(gsum, 0.0f, 1);

    // ---- layer 2 GEMMs (N padded to 16; pad weight rows are zero) ----
    long long thr2 = (long long)(NB / 16) * 1 * 32;
    gemm_wmma_bf16<<<nb(thr2, B), B, 0, stream>>>(hbf, w2lb, bl2, NCLS, xl2, NB, D1, N2PAD, 1);
    gemm_wmma_bf16<<<nb(thr2, B), B, 0, stream>>>(hbf, w2rb, br2, NCLS, xr2, NB, D1, N2PAD, 1);

    // ---- layer 2 edge passes (1 head, 10 channels) ----
    edge_score_t<1, NCLS, N2PAD><<<nb((long long)ETOT, B), B, 0, stream>>>(ei, xl2, xr2, att2, sc2, sm2);
    edge_exp_t<1>               <<<nb((long long)ETOT, B), B, 0, stream>>>(ei, sc2, sm2, dn2);
    edge_agg_t<1, NCLS, N2PAD, N2PAD><<<nb((long long)ETOT, B), B, 0, stream>>>(ei, sc2, dn2, xl2, out2);

    // ---- mean over classes, then global log-softmax over nodes ----
    node_meanv   <<<nb(NB, B), B, 0, stream>>>(out2, b2, vbuf, NB);
    reduce_max   <<<nb(NB, B), B, 0, stream>>>(vbuf, gmax, NB);
    reduce_sumexp<<<nb(NB, B), B, 0, stream>>>(vbuf, gmax, gsum, NB);
    finalize_lsm <<<nb(NB, B), B, 0, stream>>>(vbuf, gmax, gsum, out, NB);
}
